// NeuralTensorLayer_6554120094087
// MI455X (gfx1250) — matrix operand order repaired and checked by reading, not compiled, not run on hardware
//
#include <hip/hip_runtime.h>

// ---------------------------------------------------------------------------
// NeuralTensorLayer for MI455X (gfx1250), B=32768, D=K=32.
// Dominant work: [B,1024] x [1024,1056] GEMM (W3 flat, W2 flat appended as
// N-tiles 64-65), run on v_wmma_f32_16x16x32_bf16 with f32 accumulation.
// A-fragments are built on the fly: for K-step s, A[m, j] = x[m,s] * x[m,j]
// (rank-1 rescale via v_pk_mul_bf16) -- the [B,1024] outer-product matrix
// never exists in memory.
// N is padded to 72 tiles (6 zero tiles): every wave owns exactly 9 tiles.
// B-loads are software-pipelined in 3-tile chunks across k-steps, with
// sched_barrier fences so the loads stay issued ahead of the WMMAs (partial
// s_wait_loadcnt instead of 0x0).
// ---------------------------------------------------------------------------

typedef __attribute__((ext_vector_type(16))) __bf16 v16bf;
typedef __attribute__((ext_vector_type(8)))  float  v8f;

union BF16x16 { unsigned u[8]; v16bf v; };

#define NTILES 72   // 64 W3 tiles + 2 W2 tiles + 6 zero-padding tiles

__device__ __forceinline__ unsigned short f2bf(float f) {
  // round-to-nearest-even f32 -> bf16 (finite inputs)
  unsigned u = __float_as_uint(f);
  u += 0x7fffu + ((u >> 16) & 1u);
  return (unsigned short)(u >> 16);
}

// ---------------------------------------------------------------------------
// Pack kernel: W3 [32,32,32,32] f32 and W2 [32,32,32] f32 -> bf16 buffer laid
// out per (k-step s, n-tile t) in contiguous-k B-fragment order (same as A):
//   B is 32x16 bf16; lanes 0-15: K rows 0..15, lanes 16-31: K rows 16..31,
//   VGPR v holds rows (2v, 2v+1) of its half; n = lane & 15.
// Buffer layout (uint4 units): [(s*72+t)*64 + g*32 + lane], g=0 -> v0..3,
// g=1 -> v4..7. t<64: W3 (col = k*32+l); t=64,65: W2; t>=66: zeros.
// ---------------------------------------------------------------------------
__global__ void ntl_pack(const float* __restrict__ W3,
                         const float* __restrict__ W2,
                         unsigned* __restrict__ wpack) {
  const int b   = blockIdx.x;       // b = s*72 + t
  const int s   = b / NTILES;
  const int t   = b - s * NTILES;
  const int tid = threadIdx.x;      // writes one u32 (two bf16)

  const int g    = tid >> 7;            // VGPR group (0: v0-3, 1: v4-7)
  const int lane = (tid >> 2) & 31;
  const int e2   = tid & 3;             // pair index within group
  const int v    = g * 4 + e2;          // B-fragment VGPR 0..7
  const int hb   = (lane >> 4) * 16;    // K-half base
  const int n    = lane & 15;           // column
  const int kk   = hb + 2 * v;          // K rows kk, kk+1 (j index of W3)

  float f0 = 0.f, f1 = 0.f;
  if (t < 64) {
    const int k  = t >> 1;
    const int l  = (t & 1) * 16 + n;
    const int i0 = ((s * 32 + kk) * 32 + k) * 32 + l;  // W3[i=s][j=kk][k][l]
    f0 = W3[i0];
    f1 = W3[i0 + 1024];                                 // j -> j+1 stride 32*32
  } else if (t < 66) {
    const int c  = (t - 64) * 16 + n;
    const int i0 = (s * 32 + kk) * 32 + c;              // W2[i=s][j=kk][c]
    f0 = W2[i0];
    f1 = W2[i0 + 32];                                   // j -> j+1 stride 32
  }
  wpack[b * 256 + tid] = (unsigned)f2bf(f0) | ((unsigned)f2bf(f1) << 16);
}

// ---------------------------------------------------------------------------
// GEMM helpers: 3-tile chunks. load_chunk issues 6 global_load_b128;
// wmma_chunk runs 6 WMMAs against a previously loaded chunk.
// ---------------------------------------------------------------------------
__device__ __forceinline__ void load_chunk(const uint4* __restrict__ wp,
                                           int base, uint4 c[6]) {
#pragma unroll
  for (int j = 0; j < 3; ++j) {
    c[2 * j]     = wp[base + j * 64];
    c[2 * j + 1] = wp[base + j * 64 + 32];
  }
}

__device__ __forceinline__ void wmma_chunk(const uint4 c[6],
                                           const BF16x16& a0, const BF16x16& a1,
                                           v8f* A0, v8f* A1) {
#pragma unroll
  for (int j = 0; j < 3; ++j) {
    BF16x16 bb;
    bb.u[0] = c[2 * j].x;     bb.u[1] = c[2 * j].y;
    bb.u[2] = c[2 * j].z;     bb.u[3] = c[2 * j].w;
    bb.u[4] = c[2 * j + 1].x; bb.u[5] = c[2 * j + 1].y;
    bb.u[6] = c[2 * j + 1].z; bb.u[7] = c[2 * j + 1].w;
    A0[j] = __builtin_amdgcn_wmma_f32_16x16x32_bf16(
        false, a0.v, false, bb.v, (short)0, A0[j], false, false);
    A1[j] = __builtin_amdgcn_wmma_f32_16x16x32_bf16(
        false, a1.v, false, bb.v, (short)0, A1[j], false, false);
  }
}

// ---------------------------------------------------------------------------
// Per-tile epilogue: fold t3 accumulators with x[m,k] (tiles t<64) or add
// order-2 / zero-pad tiles (t>=64). LH (output column half) is a template
// parameter so part[] indexing stays compile-time constant -> registers.
// C/D layout: VGPR r, lane l -> M = r + 8*(l>=16), N = l&15.
// ---------------------------------------------------------------------------
template <int LH>
__device__ __forceinline__ void tile_reduce(int t, const v8f& A0, const v8f& A1,
                                            const float* XT, int jb,
                                            float part[2][2][8]) {
  if (t < 64) {
    const int k = t >> 1;
#pragma unroll
    for (int mt = 0; mt < 2; ++mt) {
      const float4* xp = (const float4*)(XT + k * 32 + mt * 16 + jb);
      const float4 xa = xp[0];
      const float4 xb = xp[1];
      const float xk[8] = {xa.x, xa.y, xa.z, xa.w, xb.x, xb.y, xb.z, xb.w};
      const v8f& A = mt ? A1 : A0;
#pragma unroll
      for (int r = 0; r < 8; ++r)
        part[mt][LH][r] = fmaf(A[r], xk[r], part[mt][LH][r]);
    }
  } else {
#pragma unroll
    for (int mt = 0; mt < 2; ++mt) {
      const v8f& A = mt ? A1 : A0;
#pragma unroll
      for (int r = 0; r < 8; ++r) part[mt][LH][r] += A[r];
    }
  }
}

// ---------------------------------------------------------------------------
// Main kernel: one block = 256 threads = 8 waves, 32 samples (2 M-tiles of 16).
// Each wave owns both M-tiles and exactly 9 contiguous N-tiles.
// ---------------------------------------------------------------------------
__global__ __launch_bounds__(256, 2)
void ntl_main(const float* __restrict__ X, const float* __restrict__ W1,
              const float* __restrict__ bias, const uint4* __restrict__ wpack,
              float* __restrict__ out) {
  __shared__ __align__(16) float    XT[1024];     // x transposed: XT[i*32+m]
  __shared__            unsigned  Xdup[1024];     // bf16(x[m,i]) duplicated, [i*32+m]
  __shared__            unsigned   Xbf[512];      // bf16 x, row-major packed pairs
  __shared__ __align__(16) float outAcc[1024];    // per-block output tile [m*32+l]

  const int tid = threadIdx.x;
  const int wg  = blockIdx.x;

  // ---- stage X tile: 32 samples x 32 features ----
  {
    const int m  = tid >> 3;
    const int i0 = (tid & 7) * 4;
    const float4 xv = *(const float4*)(X + (wg * 32 + m) * 32 + i0);
    XT[(i0 + 0) * 32 + m] = xv.x;
    XT[(i0 + 1) * 32 + m] = xv.y;
    XT[(i0 + 2) * 32 + m] = xv.z;
    XT[(i0 + 3) * 32 + m] = xv.w;
    Xdup[(i0 + 0) * 32 + m] = 0x00010001u * (unsigned)f2bf(xv.x);
    Xdup[(i0 + 1) * 32 + m] = 0x00010001u * (unsigned)f2bf(xv.y);
    Xdup[(i0 + 2) * 32 + m] = 0x00010001u * (unsigned)f2bf(xv.z);
    Xdup[(i0 + 3) * 32 + m] = 0x00010001u * (unsigned)f2bf(xv.w);
    Xbf[m * 16 + (i0 >> 1) + 0] = (unsigned)f2bf(xv.x) | ((unsigned)f2bf(xv.y) << 16);
    Xbf[m * 16 + (i0 >> 1) + 1] = (unsigned)f2bf(xv.z) | ((unsigned)f2bf(xv.w) << 16);
  }
  __syncthreads();

  // ---- order-1 term in f32 VALU: outAcc = bias + X @ W1 ----
  {
    const int m  = tid >> 3;
    const int l0 = (tid & 7) * 4;
    float4 a = *(const float4*)(bias + l0);
#pragma unroll 4
    for (int i = 0; i < 32; ++i) {
      const float  xv = XT[i * 32 + m];
      const float4 w  = *(const float4*)(W1 + i * 32 + l0);
      a.x = fmaf(xv, w.x, a.x);
      a.y = fmaf(xv, w.y, a.y);
      a.z = fmaf(xv, w.z, a.z);
      a.w = fmaf(xv, w.w, a.w);
    }
    *(float4*)(outAcc + m * 32 + l0) = a;
  }
  __syncthreads();

  // ---- per-wave setup ----
  const int lane = tid & 31;
  const int w    = tid >> 5;
  const int lb   = lane & 15;
  const int jb   = (lane >> 4) * 8;   // K/M sub-offset for hi lanes
  const int t0   = w * 9;             // this wave's 9 N-tiles

  // X fragments as WMMA A operands, contiguous k (lanes 0-15: K 0-15, lanes 16-31: K 16-31)
  BF16x16 xf0, xf1;
  {
    const int b0 = lb * 16 + jb;
    const int b1 = (16 + lb) * 16 + jb;
#pragma unroll
    for (int q = 0; q < 4; ++q) {
      xf0.u[q]     = Xbf[b0 + q];
      xf0.u[4 + q] = Xbf[b0 + 4 + q];
      xf1.u[q]     = Xbf[b1 + q];
      xf1.u[4 + q] = Xbf[b1 + 4 + q];
    }
  }

  const v8f vzero = {0.f, 0.f, 0.f, 0.f, 0.f, 0.f, 0.f, 0.f};
  v8f acc0[9], acc1[9];
#pragma unroll
  for (int i = 0; i < 9; ++i) { acc0[i] = vzero; acc1[i] = vzero; }

  // ---- main GEMM loop over the 32 K-steps (i index of W3) ----
  // Software pipeline (3-tile chunks, carried across iterations):
  //   [load c1] [build A] | [wmma c0] | [load c2] | [wmma c1] |
  //   [load c0 of s+1]    | [wmma c2]
  // sched_barrier(0) fences keep the loads issued ahead of the WMMAs so
  // 1-2 chunks (6-12 global_load_b128) stay in flight at all times.
  uint4 c0[6], c1[6], c2[6];
  load_chunk(wpack, t0 * 64 + lane, c0);          // s=0 chunk 0

  for (int s = 0; s < 32; ++s) {
    const int sbase = (s * NTILES + t0) * 64 + lane;
    const int snxt  = (s < 31) ? (s + 1) : 31;    // clamped prefetch (no OOB)
    const int nbase = (snxt * NTILES + t0) * 64 + lane;

    load_chunk(wpack, sbase + 192, c1);           // tiles 3-5 in flight

    // A-fragment build (VALU/DS) overlaps outstanding loads
    const unsigned xd0 = Xdup[s * 32 + lb];       // bf16(x[m,s]) duplicated
    const unsigned xd1 = Xdup[s * 32 + 16 + lb];
    BF16x16 a0, a1;
#pragma unroll
    for (int q = 0; q < 8; ++q) {
      asm("v_pk_mul_bf16 %0, %1, %2" : "=v"(a0.u[q]) : "v"(xf0.u[q]), "v"(xd0));
      asm("v_pk_mul_bf16 %0, %1, %2" : "=v"(a1.u[q]) : "v"(xf1.u[q]), "v"(xd1));
    }

    __builtin_amdgcn_sched_barrier(0);
    wmma_chunk(c0, a0, a1, acc0 + 0, acc1 + 0);   // compute tiles 0-2
    __builtin_amdgcn_sched_barrier(0);
    load_chunk(wpack, sbase + 384, c2);           // tiles 6-8 in flight
    __builtin_amdgcn_sched_barrier(0);
    wmma_chunk(c1, a0, a1, acc0 + 3, acc1 + 3);   // compute tiles 3-5
    __builtin_amdgcn_sched_barrier(0);
    load_chunk(wpack, nbase, c0);                 // next s, tiles 0-2 in flight
    __builtin_amdgcn_sched_barrier(0);
    wmma_chunk(c2, a0, a1, acc0 + 6, acc1 + 6);   // compute tiles 6-8
    __builtin_amdgcn_sched_barrier(0);
  }

  // ---- epilogue: out[m,l] += sum_k t3[m,k,l]*x[m,k]  (+ order-2 tiles) ----
  float part[2][2][8];
#pragma unroll
  for (int mt = 0; mt < 2; ++mt)
#pragma unroll
    for (int lh = 0; lh < 2; ++lh)
#pragma unroll
      for (int r = 0; r < 8; ++r) part[mt][lh][r] = 0.f;

#pragma unroll
  for (int ti = 0; ti < 9; ++ti) {
    const int t = t0 + ti;   // tile t: output cols l = (t&1)*16 + n, k = t>>1
    if ((t & 1) == 0) tile_reduce<0>(t, acc0[ti], acc1[ti], XT, jb, part);
    else              tile_reduce<1>(t, acc0[ti], acc1[ti], XT, jb, part);
  }

#pragma unroll
  for (int mt = 0; mt < 2; ++mt)
#pragma unroll
    for (int lh = 0; lh < 2; ++lh)
#pragma unroll
      for (int r = 0; r < 8; ++r)
        atomicAdd(&outAcc[(mt * 16 + jb + r) * 32 + lh * 16 + lb],
                  part[mt][lh][r]);
  __syncthreads();

  // ---- store block's 32x32 output tile ----
  *(float4*)(out + (long)wg * 1024 + tid * 4) =
      *(const float4*)(outAcc + tid * 4);
}

// ---------------------------------------------------------------------------
// Launch: pack (deterministic, every call) then main GEMM. Workspace needs
// 32*72*1024 = 2,359,296 bytes for the bf16-packed W3+W2(+padding).
// ---------------------------------------------------------------------------
extern "C" void kernel_launch(void* const* d_in, const int* in_sizes, int n_in,
                              void* d_out, int out_size, void* d_ws, size_t ws_size,
                              hipStream_t stream) {
  const float* X    = (const float*)d_in[0];
  const float* W1   = (const float*)d_in[1];
  const float* W2   = (const float*)d_in[2];
  const float* W3   = (const float*)d_in[3];
  const float* bias = (const float*)d_in[4];

  unsigned* wpack = (unsigned*)d_ws;
  const int Bn = in_sizes[0] / 32;   // 32768 samples

  ntl_pack<<<dim3(32 * NTILES), dim3(256), 0, stream>>>(W3, W2, wpack);
  ntl_main<<<dim3(Bn / 32), dim3(256), 0, stream>>>(
      X, W1, bias, (const uint4*)d_ws, (float*)d_out);
}
